// GIN_87703232184762
// MI455X (gfx1250) — matrix operand order repaired and checked by reading, not compiled, not run on hardware
//
#include <hip/hip_runtime.h>
#include <hip/hip_bf16.h>

typedef __attribute__((ext_vector_type(16))) _Float16 v16h;
typedef __attribute__((ext_vector_type(8)))  float    v8f;
typedef __attribute__((ext_vector_type(4)))  float    v4f;

#define NN    50000
#define KNB   8
#define H128  128

// ---------------------------------------------------------------------------
// GBF expansion kernels
// bond: [N,8] -> [N, 8*48], mu = linspace(0,8,48), gamma = 8/48
__global__ void k_gbf_bond(const float* __restrict__ bf, float* __restrict__ out) {
    size_t idx = (size_t)blockIdx.x * blockDim.x + threadIdx.x;
    if (idx >= (size_t)NN * 384) return;
    int i = (int)(idx / 384), rem = (int)(idx % 384);
    int j = rem / 48, s = rem % 48;
    float mu = (float)s * (8.0f / 47.0f);
    float g  = 8.0f / 48.0f;
    float d  = bf[i * 8 + j] - mu;
    out[idx] = __expf(-(d * d) / (g * g));
}
// angle: [N,8,8] -> [N, 64*6], mu = linspace(-1,1,6), gamma = 2/6
__global__ void k_gbf_angle(const float* __restrict__ af, float* __restrict__ out) {
    size_t idx = (size_t)blockIdx.x * blockDim.x + threadIdx.x;
    if (idx >= (size_t)NN * 384) return;
    int i = (int)(idx / 384), rem = (int)(idx % 384);
    int pr = rem / 6, s = rem % 6;
    float mu = -1.0f + (float)s * (2.0f / 5.0f);
    float g  = 2.0f / 6.0f;
    float d  = af[i * 64 + pr] - mu;
    out[idx] = __expf(-(d * d) / (g * g));
}

// ---------------------------------------------------------------------------
// Weight convert: W[k,n] fp32 -> Wt[n,k] f16 (row of Wt is the B-fragment source)
__global__ void k_wconv(const float* __restrict__ W, _Float16* __restrict__ Wt,
                        int K, int Nout) {
    int idx = blockIdx.x * blockDim.x + threadIdx.x;
    if (idx >= K * Nout) return;
    int n = idx / K, k = idx % K;
    Wt[idx] = (_Float16)W[k * Nout + n];
}

// ---------------------------------------------------------------------------
// WMMA GEMM: out[M, NT*16](ld=ldOut) = act( A[M,K] @ W[K, NT*16] + bias )
//  - A fp32, converted to f16 fragments on the fly (one A-frag load per K-step,
//    reused across all NT column tiles -> minimal HBM traffic on the big operand)
//  - Weights pre-transposed f16 and staged once per block into LDS; B-fragments
//    are contiguous 32B ds reads
//  - fp32 accumulate; 16x16 C/D tiles per ISA VGPR layout
template <int NT>
__global__ void k_gemm_nt(const float* __restrict__ A, const _Float16* __restrict__ Wt,
                          const float* __restrict__ bias, float* __restrict__ out,
                          int M, int K, int ldOut, int relu) {
    extern __shared__ _Float16 lds[];
    // cooperative stage of the whole (NT*16 x K) f16 weight block
    const int words = (NT * 16 * K) >> 1;
    for (int w = threadIdx.x; w < words; w += blockDim.x)
        ((unsigned int*)lds)[w] = ((const unsigned int*)Wt)[w];
    __syncthreads();

    const int wave = threadIdx.x >> 5;
    const int lane = threadIdx.x & 31;
    const int hi   = lane >> 4;          // half-wave select
    const int l16  = lane & 15;
    const int m0   = (blockIdx.x * 8 + wave) << 4;

    int mrow = m0 + l16; if (mrow >= M) mrow = M - 1;   // clamp; stores guarded
    const float* ap = A + (size_t)mrow * K + hi * 16;   // A-frag K={kb..kb+15}

    v8f acc[NT];
    #pragma unroll
    for (int t = 0; t < NT; ++t) acc[t] = (v8f){0.f,0.f,0.f,0.f,0.f,0.f,0.f,0.f};

    for (int k = 0; k < K; k += 32) {
        __builtin_prefetch(ap + k + 32, 0, 0);          // global_prefetch_b8
        v4f a0 = *(const v4f*)(ap + k);
        v4f a1 = *(const v4f*)(ap + k + 4);
        v4f a2 = *(const v4f*)(ap + k + 8);
        v4f a3 = *(const v4f*)(ap + k + 12);
        v16h af;
        #pragma unroll
        for (int i = 0; i < 4; ++i) {
            af[i]      = (_Float16)a0[i];
            af[4 + i]  = (_Float16)a1[i];
            af[8 + i]  = (_Float16)a2[i];
            af[12 + i] = (_Float16)a3[i];
        }
        #pragma unroll
        for (int t = 0; t < NT; ++t) {
            const _Float16* bp = lds + (size_t)(t * 16 + l16) * K + hi * 16 + k;
            v16h bf = *(const v16h*)bp;                 // ds_load_b128 x2
            acc[t] = __builtin_amdgcn_wmma_f32_16x16x32_f16(false, af, false, bf,
                                                            (short)0, acc[t],
                                                            false, false);
        }
    }

    #pragma unroll
    for (int t = 0; t < NT; ++t) {
        const int ncol = t * 16 + l16;
        const float bv = bias ? bias[ncol] : 0.0f;
        #pragma unroll
        for (int v = 0; v < 8; ++v) {
            int r = m0 + v + hi * 8;                    // C/D VGPR layout
            if (r < M) {
                float val = acc[t][v] + bv;
                if (relu) val = fmaxf(val, 0.0f);
                out[(size_t)r * ldOut + ncol] = val;
            }
        }
    }
}

// ---------------------------------------------------------------------------
// GIN aggregation (128-wide): u pre-initialized to t (memcpy), then scatter-add
__global__ void k_scatter128(const float* __restrict__ t, float* __restrict__ u,
                             const int* __restrict__ nbr) {
    size_t idx = (size_t)blockIdx.x * blockDim.x + threadIdx.x;
    if (idx >= (size_t)NN * KNB * H128) return;
    int c = (int)(idx & 127);
    int e = (int)(idx >> 7);
    int i = e >> 3;                 // source node (edges are node-major)
    int d = nbr[e];                 // destination node
    atomicAdd(&u[(size_t)d * H128 + c], t[(size_t)i * H128 + c]);
}

__global__ void k_bias_relu128(float* __restrict__ x, const float* __restrict__ b) {
    size_t idx = (size_t)blockIdx.x * blockDim.x + threadIdx.x;
    if (idx >= (size_t)NN * H128) return;
    x[idx] = fmaxf(x[idx] + b[idx & 127], 0.0f);
}

// ---------------------------------------------------------------------------
// BatchNorm (training-mode batch stats, biased variance)
__global__ void k_zero(float* p, int n) {
    int i = blockIdx.x * blockDim.x + threadIdx.x;
    if (i < n) p[i] = 0.0f;
}
__global__ void k_bnstats(const float* __restrict__ x, float* __restrict__ stats) {
    int c = threadIdx.x;            // blockDim.x == 128
    float s = 0.0f, s2 = 0.0f;
    for (int r = blockIdx.x; r < NN; r += gridDim.x) {
        float v = x[(size_t)r * H128 + c];
        s += v; s2 += v * v;
    }
    atomicAdd(&stats[c], s);
    atomicAdd(&stats[H128 + c], s2);
}
__global__ void k_bnapply(float* __restrict__ x, const float* __restrict__ stats,
                          const float* __restrict__ g, const float* __restrict__ be) {
    size_t idx = (size_t)blockIdx.x * blockDim.x + threadIdx.x;
    if (idx >= (size_t)NN * H128) return;
    int c = (int)(idx & 127);
    float m   = stats[c] * (1.0f / (float)NN);
    float var = stats[H128 + c] * (1.0f / (float)NN) - m * m;
    x[idx] = (x[idx] - m) * rsqrtf(var + 1e-5f) * g[c] + be[c];
}

// ---------------------------------------------------------------------------
// Final classifier: [N,64] @ [64,2] + b  (Nout=2 not WMMA-conforming)
__global__ void k_fcout(const float* __restrict__ x, const float* __restrict__ W,
                        const float* __restrict__ b, float* __restrict__ out) {
    int r = blockIdx.x * blockDim.x + threadIdx.x;
    if (r >= NN) return;
    float a0 = b[0], a1 = b[1];
    const float* xr = x + (size_t)r * 64;
    #pragma unroll
    for (int k = 0; k < 64; ++k) {
        a0 += xr[k] * W[k * 2 + 0];
        a1 += xr[k] * W[k * 2 + 1];
    }
    out[r * 2 + 0] = a0;
    out[r * 2 + 1] = a1;
}

// ---------------------------------------------------------------------------
extern "C" void kernel_launch(void* const* d_in, const int* in_sizes, int n_in,
                              void* d_out, int out_size, void* d_ws, size_t ws_size,
                              hipStream_t stream) {
    (void)in_sizes; (void)n_in; (void)out_size; (void)ws_size;

    const float* bond_fea  = (const float*)d_in[0];
    const float* angle_fea = (const float*)d_in[1];
    const int*   nbr_idx   = (const int*)d_in[3];
    auto P = [&](int i) { return (const float*)d_in[i]; };
    // params flattened in sorted-key pytree order (offset 5):
    // 5,6: bn1_angle(g,b)  7,8: bn1_bond(g,b)
    // 9..20: em_angle[2] x (W1,b1,W2,b2,g,be)   21..32: em_bond[2]
    // 33,34: fc(W,b)  35,36: fc1_angle  37,38: fc1_bond
    // 39,40: fc2_angle  41,42: fc2_bond  43..46: mlp[2] x (W,b)
    // 47..50: nn1_angle(W1,b1,W2,b2)  51..54: nn1_bond

    // workspace partition (fp32 elements)
    float* ws    = (float*)d_ws;
    float* f384  = ws;                       // 50000*384 = 19,200,000
    float* t     = f384 + 19200000;          // 50000*128
    float* u     = t    + 6400000;           // 50000*128
    float* actB  = u    + 6400000;
    float* actA  = actB + 6400000;
    float* crys  = actA + 6400000;           // 50000*64
    float* crys2 = crys + 3200000;           // 50000*64
    float* stats = crys2 + 3200000;          // 256
    _Float16* wbuf = (_Float16*)(stats + 256); // up to 384*128 halves (32B-aligned)

    const size_t NE   = (size_t)NN * H128;            // 6.4M
    const int    B1   = (int)((NE + 255) / 256);
    const int    B384 = (int)(((size_t)NN * 384 + 255) / 256);
    const int    BSC  = (int)(((size_t)NN * KNB * H128 + 255) / 256);

    auto GEMM = [&](const float* A, const float* W, const float* bias, float* out,
                    int K, int Nout, int ldOut, bool relu) {
        int wn = K * Nout;
        k_wconv<<<(wn + 255) / 256, 256, 0, stream>>>(W, wbuf, K, Nout);
        dim3 grid((NN + 127) / 128);
        size_t shmem = (size_t)Nout * K * sizeof(_Float16);   // staged weights
        int r = relu ? 1 : 0;
        switch (Nout) {
        case 128: k_gemm_nt<8><<<grid, 256, shmem, stream>>>(A, wbuf, bias, out, NN, K, ldOut, r); break;
        case 64:  k_gemm_nt<4><<<grid, 256, shmem, stream>>>(A, wbuf, bias, out, NN, K, ldOut, r); break;
        default:  k_gemm_nt<2><<<grid, 256, shmem, stream>>>(A, wbuf, bias, out, NN, K, ldOut, r); break;
        }
    };
    auto AGG = [&](const float* tin, float* uout) {   // uout = tin + scatter_sum(tin)
        hipMemcpyAsync(uout, tin, NE * sizeof(float), hipMemcpyDeviceToDevice, stream);
        k_scatter128<<<BSC, 256, 0, stream>>>(tin, uout, nbr_idx);
    };
    auto BN = [&](float* x, const float* g, const float* be) {
        k_zero<<<1, 256, 0, stream>>>(stats, 256);
        k_bnstats<<<512, 128, 0, stream>>>(x, stats);
        k_bnapply<<<B1, 256, 0, stream>>>(x, stats, g, be);
    };
    // One GIN conv step. Uses agg(x)@W1 == agg(x@W1) to aggregate at width 128.
    auto CONV = [&](float* act, const float* W1, const float* b1,
                    const float* W2, const float* b2,
                    const float* g, const float* be, const float* Ain, int Kin) {
        GEMM(Ain, W1, nullptr, t, Kin, H128, H128, false);   // t = Ain @ W1
        AGG(t, u);                                           // u = agg(t)
        k_bias_relu128<<<B1, 256, 0, stream>>>(u, b1);       // u = relu(u + b1)
        GEMM(u, W2, b2, act, H128, H128, H128, true);        // act = relu(u@W2 + b2)
        BN(act, g, be);
    };

    // ---------------- bond branch ----------------
    k_gbf_bond<<<B384, 256, 0, stream>>>(bond_fea, f384);
    CONV(actB, P(51), P(52), P(53), P(54), P(7), P(8), f384, 384);    // nn1_bond + bn1_bond
    CONV(actB, P(21), P(22), P(23), P(24), P(25), P(26), actB, H128); // em_bond[0]
    CONV(actB, P(27), P(28), P(29), P(30), P(31), P(32), actB, H128); // em_bond[1]
    GEMM(actB, P(37), P(38), t, H128, 64, 64, false);    // fc1_bond -> [N,64] (in t)
    GEMM(t,    P(41), P(42), crys, 64, 32, 64, false);   // fc2_bond -> crys[:, 0:32]

    // ---------------- angle branch ----------------
    k_gbf_angle<<<B384, 256, 0, stream>>>(angle_fea, f384);
    CONV(actA, P(47), P(48), P(49), P(50), P(5), P(6), f384, 384);    // nn1_angle + bn1_angle
    CONV(actA, P(9),  P(10), P(11), P(12), P(13), P(14), actA, H128); // em_angle[0]
    CONV(actA, P(15), P(16), P(17), P(18), P(19), P(20), actA, H128); // em_angle[1]
    GEMM(actA, P(35), P(36), t, H128, 64, 64, false);        // fc1_angle
    GEMM(t,    P(39), P(40), crys + 32, 64, 32, 64, false);  // fc2_angle -> crys[:,32:64]

    // ---------------- head ----------------
    GEMM(crys,  P(43), P(44), crys2, 64, 64, 64, false);  // mlp[0]
    GEMM(crys2, P(45), P(46), crys,  64, 64, 64, false);  // mlp[1]
    k_fcout<<<(NN + 255) / 256, 256, 0, stream>>>(crys, P(33), P(34), (float*)d_out);
}
